// AttentionLayer_82635170775524
// MI455X (gfx1250) — matrix operand order repaired and checked by reading, not compiled, not run on hardware
//
#include <hip/hip_runtime.h>
#include <hip/hip_bf16.h>

// ---------------------------------------------------------------------------
// attention = softmax(x1 @ W); out = einsum('btd,bsd->bts', attention, x0)
// B=8, T=2048, D=1024.  bf16 WMMA (v_wmma_f32_16x16x32_bf16) for both GEMMs,
// f32 accumulation, f32 softmax.  GEMM1 fused with softmax through LDS.
// ---------------------------------------------------------------------------

typedef unsigned short u16;
typedef __bf16  bf16x8  __attribute__((ext_vector_type(8)));
typedef __bf16  v16bf   __attribute__((ext_vector_type(16)));
typedef float   v8f     __attribute__((ext_vector_type(8)));

#define BATCH 8
#define TT    2048
#define DD    1024
#define BT    (BATCH * TT)   // 16384 rows for GEMM1

__device__ __forceinline__ u16 f32_to_bf16_rne(float f) {
    union { float f; unsigned u; } v; v.f = f;
    unsigned u = v.u;
    unsigned r = u + 0x7FFFu + ((u >> 16) & 1u);   // round-to-nearest-even
    return (u16)(r >> 16);
}

// ---- elementwise f32 -> bf16 convert (vectorized, grid-stride) -------------
__global__ __launch_bounds__(256) void k_cvt_bf16(const float* __restrict__ in,
                                                  u16* __restrict__ out, int n4) {
    int i = blockIdx.x * blockDim.x + threadIdx.x;
    int stride = gridDim.x * blockDim.x;
    const float4* in4 = (const float4*)in;
    for (; i < n4; i += stride) {
        float4 v = in4[i];
        ushort4 o;
        o.x = f32_to_bf16_rne(v.x);
        o.y = f32_to_bf16_rne(v.y);
        o.z = f32_to_bf16_rne(v.z);
        o.w = f32_to_bf16_rne(v.w);
        ((ushort4*)out)[i] = o;
    }
}

// ---- W (DxD, row-major [k][n]) -> Wt bf16 [n][k] ---------------------------
__global__ __launch_bounds__(256) void k_transpose_W(const float* __restrict__ W,
                                                     u16* __restrict__ Wt) {
    int idx = blockIdx.x * blockDim.x + threadIdx.x;   // coalesced read
    if (idx < DD * DD) {
        int k = idx >> 10;
        int n = idx & (DD - 1);
        Wt[n * DD + k] = f32_to_bf16_rne(W[idx]);
    }
}

// ---- fragment loaders (16-bit A 16x32 / B 32x16 VGPR layouts, ISA 7.12.2) --
// A: lane m = lane&15; lanes<16 hold K[0..7]+K[16..23], lanes>=16 K[8..15]+K[24..31]
__device__ __forceinline__ v16bf load_frag_a(const u16* __restrict__ rowBase, int k) {
    int lane  = threadIdx.x & 31;
    int koff  = (lane >> 4) * 8;
    bf16x8 lo = *(const bf16x8*)(rowBase + k + koff);
    bf16x8 hi = *(const bf16x8*)(rowBase + k + 16 + koff);
    return __builtin_shufflevector(lo, hi, 0,1,2,3,4,5,6,7,8,9,10,11,12,13,14,15);
}
// B: lane n = lane&15; lanes<16 hold K[0..7]+K[16..23], lanes>=16 K[8..15]+K[24..31]
__device__ __forceinline__ v16bf load_frag_b(const u16* __restrict__ colBase, int k) {
    int lane  = threadIdx.x & 31;
    int koff  = (lane >> 4) * 8;
    bf16x8 lo = *(const bf16x8*)(colBase + k + koff);
    bf16x8 hi = *(const bf16x8*)(colBase + k + 16 + koff);
    return __builtin_shufflevector(lo, hi, 0,1,2,3,4,5,6,7,8,9,10,11,12,13,14,15);
}

// ---------------------------------------------------------------------------
// GEMM1 + softmax fused.  One block = 16 rows x full 1024 cols of scores.
// 8 waves; wave w owns columns [w*128, w*128+128) = 8 WMMA N-tiles.
// ---------------------------------------------------------------------------
__global__ __launch_bounds__(256, 1)
void k_gemm1_softmax(const u16* __restrict__ x1bf, const u16* __restrict__ Wt,
                     float* __restrict__ attnF32, u16* __restrict__ attnBf) {
    __shared__ float sC[16 * DD];       // 64 KB score stripe
    __shared__ float sRed[16 * 16];
    __shared__ float sRowMax[16];
    __shared__ float sRowSum[16];

    const int tid  = threadIdx.x;
    const int wave = tid >> 5;
    const int lane = tid & 31;
    const int laneM = lane & 15;
    const int r0 = blockIdx.x * 16;

    const u16* aRow = x1bf + (size_t)(r0 + laneM) * DD;
    const u16* bRow[8];
#pragma unroll
    for (int nt = 0; nt < 8; ++nt)
        bRow[nt] = Wt + (size_t)(wave * 128 + nt * 16 + laneM) * DD;

    v8f acc[8];
#pragma unroll
    for (int nt = 0; nt < 8; ++nt) acc[nt] = (v8f){0,0,0,0,0,0,0,0};

    for (int k = 0; k < DD; k += 32) {
        v16bf av = load_frag_a(aRow, k);
#pragma unroll
        for (int nt = 0; nt < 8; ++nt) {
            v16bf bv = load_frag_b(bRow[nt], k);
            acc[nt] = __builtin_amdgcn_wmma_f32_16x16x32_bf16(
                false, av, false, bv, (short)0, acc[nt], false, false);
        }
    }

    // park scores in LDS (C layout: VGPR j -> row j + (lane>=16)*8, col lane&15)
    const int rBase = (lane >> 4) * 8;
#pragma unroll
    for (int nt = 0; nt < 8; ++nt)
#pragma unroll
        for (int j = 0; j < 8; ++j)
            sC[(rBase + j) * DD + wave * 128 + nt * 16 + laneM] = acc[nt][j];
    __syncthreads();

    // ---- softmax over each 1024-wide row: 16 threads/row, 64 elems/thread --
    const int row = tid >> 4;
    const int seg = tid & 15;
    float* rp = sC + row * DD + seg * 64;

    float m = -3.4e38f;
#pragma unroll 8
    for (int i = 0; i < 64; ++i) m = fmaxf(m, rp[i]);
    sRed[row * 16 + seg] = m;
    __syncthreads();
    if (seg == 0) {
        float mm = sRed[row * 16];
#pragma unroll
        for (int i = 1; i < 16; ++i) mm = fmaxf(mm, sRed[row * 16 + i]);
        sRowMax[row] = mm;
    }
    __syncthreads();

    const float rm = sRowMax[row];
    float s = 0.f;
#pragma unroll 8
    for (int i = 0; i < 64; ++i) {
        float e = __expf(rp[i] - rm);
        rp[i] = e;
        s += e;
    }
    sRed[row * 16 + seg] = s;
    __syncthreads();
    if (seg == 0) {
        float ss = 0.f;
#pragma unroll
        for (int i = 0; i < 16; ++i) ss += sRed[row * 16 + i];
        sRowSum[row] = ss;
    }
    __syncthreads();

    const float inv = 1.0f / sRowSum[row];
    const size_t base = (size_t)(r0 + row) * DD + seg * 64;
#pragma unroll 8
    for (int i = 0; i < 64; ++i) {
        float a = rp[i] * inv;
        attnF32[base + i] = a;
        attnBf[base + i]  = f32_to_bf16_rne(a);
    }
}

// ---------------------------------------------------------------------------
// GEMM2 (NT): out[b,t,s] = sum_d attn[b,t,d] * x0[b,s,d]
// Block = 128x128 C tile; 8 waves in 4(M) x 2(N); wave = 2x4 WMMA tiles.
// Both operands K-contiguous -> b128 fragment loads straight from L2.
// ---------------------------------------------------------------------------
__global__ __launch_bounds__(256, 2)
void k_gemm2(const u16* __restrict__ attnBf, const u16* __restrict__ x0bf,
             float* __restrict__ out) {
    const int tid  = threadIdx.x;
    const int wave = tid >> 5;
    const int lane = tid & 31;
    const int laneM = lane & 15;
    const int wM = wave & 3;          // 4 wave-rows
    const int wN = wave >> 2;         // 2 wave-cols
    const int b  = blockIdx.z;
    const int m0 = blockIdx.y * 128;
    const int n0 = blockIdx.x * 128;

    const u16* aRow[2];
    const u16* bRow[4];
#pragma unroll
    for (int mt = 0; mt < 2; ++mt)
        aRow[mt] = attnBf + ((size_t)b * TT + m0 + wM * 32 + mt * 16 + laneM) * DD;
#pragma unroll
    for (int nt = 0; nt < 4; ++nt)
        bRow[nt] = x0bf + ((size_t)b * TT + n0 + wN * 64 + nt * 16 + laneM) * DD;

    v8f acc[2][4];
#pragma unroll
    for (int mt = 0; mt < 2; ++mt)
#pragma unroll
        for (int nt = 0; nt < 4; ++nt) acc[mt][nt] = (v8f){0,0,0,0,0,0,0,0};

    for (int k = 0; k < DD; k += 32) {
        v16bf av[2], bv[4];
#pragma unroll
        for (int mt = 0; mt < 2; ++mt) av[mt] = load_frag_a(aRow[mt], k);
#pragma unroll
        for (int nt = 0; nt < 4; ++nt) bv[nt] = load_frag_b(bRow[nt], k);
#pragma unroll
        for (int mt = 0; mt < 2; ++mt)
#pragma unroll
            for (int nt = 0; nt < 4; ++nt)
                acc[mt][nt] = __builtin_amdgcn_wmma_f32_16x16x32_bf16(
                    false, av[mt], false, bv[nt], (short)0, acc[mt][nt], false, false);
    }

    const int rBase = (lane >> 4) * 8;
    float* outB = out + (size_t)b * TT * TT;
#pragma unroll
    for (int mt = 0; mt < 2; ++mt)
#pragma unroll
        for (int nt = 0; nt < 4; ++nt) {
            const int col = n0 + wN * 64 + nt * 16 + laneM;
#pragma unroll
            for (int j = 0; j < 8; ++j) {
                const int rowi = m0 + wM * 32 + mt * 16 + rBase + j;
                outB[(size_t)rowi * TT + col] = acc[mt][nt][j];
            }
        }
}

// ---------------------------------------------------------------------------
extern "C" void kernel_launch(void* const* d_in, const int* in_sizes, int n_in,
                              void* d_out, int out_size, void* d_ws, size_t ws_size,
                              hipStream_t stream) {
    const float* x0 = (const float*)d_in[0];   // [B,T,D]
    const float* x1 = (const float*)d_in[1];   // [B,T,D]
    const float* W  = (const float*)d_in[2];   // [D,D]

    // d_out: weighted_sum [B,T,T] f32 then attention [B,T,D] f32
    float* outWS   = (float*)d_out;
    float* attnF32 = (float*)d_out + (size_t)BATCH * TT * TT;

    // workspace layout (bf16 = u16)
    const size_t N_X = (size_t)BT * DD;              // 16,777,216 elems
    u16* x1bf   = (u16*)d_ws;
    u16* x0bf   = x1bf + N_X;
    u16* attnBf = x0bf + N_X;
    u16* Wt     = attnBf + N_X;                      // +2MB; total ~101MB

    k_cvt_bf16<<<4096, 256, 0, stream>>>(x1, x1bf, (int)(N_X / 4));
    k_cvt_bf16<<<4096, 256, 0, stream>>>(x0, x0bf, (int)(N_X / 4));
    k_transpose_W<<<(DD * DD) / 256, 256, 0, stream>>>(W, Wt);

    k_gemm1_softmax<<<BT / 16, 256, 0, stream>>>(x1bf, Wt, attnF32, attnBf);

    dim3 g2(TT / 128, TT / 128, BATCH);
    k_gemm2<<<g2, 256, 0, stream>>>(attnBf, x0bf, outWS);
}
